// ShiftWindAttention_64768106824266
// MI455X (gfx1250) — matrix operand order repaired and checked by reading, not compiled, not run on hardware
//
#include <hip/hip_runtime.h>
#include <hip/hip_bf16.h>

// ---------------------------------------------------------------------------
// Shifted-window attention (Swin-style) for MI455X / gfx1250, wave32.
// Pipeline: [qkv GEMM bf16-WMMA] -> [windowed attention bf16-WMMA] ->
//           [out-proj GEMM bf16-WMMA + bias].
// All matmuls run on v_wmma_f32_16x16x32_bf16 (fp32 accumulate).
// bf16-source tiles are staged into LDS with GLOBAL_LOAD_ASYNC_TO_LDS_B128
// (ASYNCcnt path) issued via inline asm per CDNA5 ISA.
// Workspace: qkv bf16 (65536x3072) + attn-out bf16 (65536x1024) = ~537 MB.
// ---------------------------------------------------------------------------

typedef __bf16 bf16;
typedef __attribute__((ext_vector_type(16))) __bf16 v16bf;
typedef __attribute__((ext_vector_type(8)))  float  v8f;

#define WMMA_BF16(A_, B_, C_) \
    __builtin_amdgcn_wmma_f32_16x16x32_bf16(false, (A_), false, (B_), (short)0, (C_), false, false)

// ---- async global->LDS 16B copy (gfx1250 GLOBAL_LOAD_ASYNC_TO_LDS_B128) ----
// VGLOBAL encoding: VDST = LDS byte address VGPR, VADDR = 64-bit global addr.
// Generic LDS pointer low 32 bits == LDS byte offset (flat aperture rule).
__device__ __forceinline__ void async_cp16(void* lds_dst, const void* gsrc) {
    const unsigned int       loff  = (unsigned int)(unsigned long long)lds_dst;
    const unsigned long long gaddr = (unsigned long long)gsrc;
    asm volatile("global_load_async_to_lds_b128 %0, %1, off"
                 :: "v"(loff), "v"(gaddr)
                 : "memory");
}
#define ASYNC_WAIT() asm volatile("s_wait_asynccnt 0x0" ::: "memory")

// A-matrix fragment (16x32 bf16). Contiguous k order used here (B matches): lane holds row M = lane&15;
// VGPR j holds K pair k = (lane>=16?16:0) + 2*j.
template <int STRIDE>
__device__ __forceinline__ v16bf load_a_frag(const bf16* lds, int rowbase, int lane) {
    const int r     = rowbase + (lane & 15);
    const int khalf = (lane >> 4) << 4;
    v16bf a;
#pragma unroll
    for (int j = 0; j < 8; ++j) {
        const int k = khalf + (j << 1);
        a[2 * j]     = lds[r * STRIDE + k];
        a[2 * j + 1] = lds[r * STRIDE + k + 1];
    }
    return a;
}

// B-matrix fragment (32x16 bf16, row-major [k][n] source).
template <int STRIDE>
__device__ __forceinline__ v16bf load_b_frag(const bf16* lds, int colbase, int lane) {
    const int n     = colbase + (lane & 15);
    const int khalf = (lane >> 4) << 4;
    v16bf b;
#pragma unroll
    for (int j = 0; j < 8; ++j) {
        b[2 * j]     = lds[(khalf + 2 * j)     * STRIDE + n];
        b[2 * j + 1] = lds[(khalf + 2 * j + 1) * STRIDE + n];
    }
    return b;
}

// B fragment gathered from a row-major [n][k] tile (B = K^T for Q.K^T).
template <int STRIDE>
__device__ __forceinline__ v16bf load_bT_frag(const bf16* lds, int colbase, int kbase, int lane) {
    const int n     = colbase + (lane & 15);
    const int khalf = kbase + ((lane >> 4) << 4);
    v16bf b;
#pragma unroll
    for (int j = 0; j < 8; ++j) {
        b[2 * j]     = lds[n * STRIDE + khalf + 2 * j];
        b[2 * j + 1] = lds[n * STRIDE + khalf + 2 * j + 1];
    }
    return b;
}

// ---------------------------------------------------------------------------
// 128x64-tile GEMM: C[MxN] = A[MxK] * B[KxN] (+bias), K-step 32.
// Block = 128 threads = 4 waves; each wave owns a 32x64 strip (8 WMMA tiles),
// reusing 4 B-fragments across 2 A-strips.
// ---------------------------------------------------------------------------
template <bool A_BF16, bool OUT_F32>
__global__ __launch_bounds__(128) void gemm_wmma_kernel(
    const void* __restrict__ A, const float* __restrict__ Bw,
    const float* __restrict__ bias, void* __restrict__ C,
    int M, int N, int K)
{
    const int tid  = threadIdx.x;
    const int wv   = tid >> 5;
    const int lane = tid & 31;
    const int row0 = blockIdx.y * 128;
    const int col0 = blockIdx.x * 64;

    __shared__ bf16 As[128 * 40];  // 128x32 tile, padded stride (80 B rows)
    __shared__ bf16 Bs[32 * 72];   // 32x64 tile, padded stride

    v8f acc[2][4];
    const v8f vzero = {};
#pragma unroll
    for (int s = 0; s < 2; ++s)
#pragma unroll
        for (int t = 0; t < 4; ++t) acc[s][t] = vzero;

    for (int k0 = 0; k0 < K; k0 += 32) {
        // ---- stage A tile (128x32): thread t owns row t ----
        if constexpr (A_BF16) {
            const bf16* src = (const bf16*)A + (size_t)(row0 + tid) * K + k0;
#pragma unroll
            for (int u = 0; u < 4; ++u)            // 4 x 16B async copies
                async_cp16(&As[tid * 40 + u * 8], src + u * 8);
        } else {
            const float* src = (const float*)A + (size_t)(row0 + tid) * K + k0;
#pragma unroll
            for (int c = 0; c < 32; ++c) As[tid * 40 + c] = (bf16)src[c];
        }
        // ---- stage B tile (32x64), fp32 weights -> bf16 ----
#pragma unroll
        for (int e = 0; e < 16; ++e) {
            const int idx = tid * 16 + e;
            const int r = idx >> 6, c = idx & 63;
            Bs[r * 72 + c] = (bf16)Bw[(size_t)(k0 + r) * N + col0 + c];
        }
        // prefetch next B K-tile into GL2 (global_prefetch_b8)
        if (k0 + 32 < K)
            __builtin_prefetch(&Bw[(size_t)(k0 + 32 + (tid >> 2)) * N + col0 + (tid & 3) * 16], 0, 1);
        if constexpr (A_BF16) ASYNC_WAIT();
        __syncthreads();

        const v16bf a0 = load_a_frag<40>(As, wv * 32,      lane);
        const v16bf a1 = load_a_frag<40>(As, wv * 32 + 16, lane);
#pragma unroll
        for (int t = 0; t < 4; ++t) {
            const v16bf bfrag = load_b_frag<72>(Bs, t * 16, lane);
            acc[0][t] = WMMA_BF16(a0, bfrag, acc[0][t]);
            acc[1][t] = WMMA_BF16(a1, bfrag, acc[1][t]);
        }
        __syncthreads();
    }

    // ---- epilogue: C/D layout row = j + 8*(lane>=16), col = lane&15 ----
#pragma unroll
    for (int s = 0; s < 2; ++s) {
#pragma unroll
        for (int t = 0; t < 4; ++t) {
#pragma unroll
            for (int j = 0; j < 8; ++j) {
                const int r = row0 + wv * 32 + s * 16 + j + ((lane >> 4) << 3);
                const int c = col0 + t * 16 + (lane & 15);
                const float v = acc[s][t][j];
                if constexpr (OUT_F32)
                    ((float*)C)[(size_t)r * N + c] = v + (bias ? bias[c] : 0.0f);
                else
                    ((bf16*)C)[(size_t)r * N + c] = (bf16)v;
            }
        }
    }
}

// region label for the shift-4 attention mask (P=32, WIN=8, shift=4)
__device__ __forceinline__ int swin_lab(int t) { return t < 24 ? 0 : (t < 28 ? 1 : 2); }

// ---------------------------------------------------------------------------
// Windowed attention: one block per (window, head). 128 threads = 4 waves.
// Cyclic shift handled purely by index math on load/store.
// ---------------------------------------------------------------------------
__global__ __launch_bounds__(128) void swin_attn_kernel(
    const bf16* __restrict__ qkv,          // (B*1024) x 3072 : [q|k|v][head][d]
    const float* __restrict__ bias_table,  // 225 x 16
    bf16* __restrict__ o)                  // (B*1024) x 1024 : [head][d]
{
    const int tid  = threadIdx.x;
    const int wv   = tid >> 5;
    const int lane = tid & 31;
    const int b    = blockIdx.x >> 4;
    const int w    = blockIdx.x & 15;
    const int head = blockIdx.y;
    const int shift = (head >= 8) ? 4 : 0;
    const int wy = w >> 2, wx = w & 3;

    __shared__ bf16  Qs[64 * 72];
    __shared__ bf16  Ks[64 * 72];
    __shared__ bf16  Vs[64 * 72];
    __shared__ float Sf[64 * 66];
    __shared__ bf16  Ps[64 * 72];

    // ---- Q/K/V 64x64 bf16 tiles via async global->LDS (12 x 16B / thread) ----
    {
        const int i  = tid >> 1;
        const int d0 = (tid & 1) * 32;
        const int r = i >> 3, c = i & 7;
        const int y = (wy * 8 + r + shift) & 31;
        const int x = (wx * 8 + c + shift) & 31;
        const size_t base = ((size_t)b * 1024 + y * 32 + x) * 3072 + (size_t)head * 64 + d0;
#pragma unroll
        for (int u = 0; u < 4; ++u) {
            async_cp16(&Qs[i * 72 + d0 + u * 8], qkv + base + u * 8);
            async_cp16(&Ks[i * 72 + d0 + u * 8], qkv + base + 1024 + u * 8);
            async_cp16(&Vs[i * 72 + d0 + u * 8], qkv + base + 2048 + u * 8);
        }
        ASYNC_WAIT();
    }
    __syncthreads();

    // ---- S = Q.K^T with fused scale + rel-bias + shift-mask epilogue ----
    const float scale = 0.125f;  // DH^-0.5
    const v16bf a0 = load_a_frag<72>(Qs,      wv * 16, lane);
    const v16bf a1 = load_a_frag<72>(Qs + 32, wv * 16, lane);

    // hoist row-side descriptors (8 per lane) out of the tile loop
    int igr[8], ri0[8], ci0[8], regi[8];
#pragma unroll
    for (int j = 0; j < 8; ++j) {
        const int ig = wv * 16 + j + ((lane >> 4) << 3);
        igr[j] = ig;
        ri0[j] = ig >> 3;
        ci0[j] = ig & 7;
        regi[j] = swin_lab(wy * 8 + ri0[j]) * 3 + swin_lab(wx * 8 + ci0[j]);
    }

#pragma unroll
    for (int t = 0; t < 4; ++t) {
        v8f acc = {};
        const v16bf bk0 = load_bT_frag<72>(Ks, t * 16, 0,  lane);
        const v16bf bk1 = load_bT_frag<72>(Ks, t * 16, 32, lane);
        acc = WMMA_BF16(a0, bk0, acc);
        acc = WMMA_BF16(a1, bk1, acc);

        const int jg   = t * 16 + (lane & 15);
        const int rj   = jg >> 3, cj = jg & 7;
        const int regj = swin_lab(wy * 8 + rj) * 3 + swin_lab(wx * 8 + cj);
#pragma unroll
        for (int j = 0; j < 8; ++j) {
            float s = acc[j] * scale
                    + bias_table[((ri0[j] - rj + 7) * 15 + (ci0[j] - cj + 7)) * 16 + head];
            if (shift && (regi[j] != regj)) s -= 1.0e9f;
            Sf[igr[j] * 66 + jg] = s;
        }
    }
    __syncthreads();

    // ---- row softmax (fp32), result as bf16 P ----
    if (tid < 64) {
        float m = -1.0e30f;
        for (int j = 0; j < 64; ++j) m = fmaxf(m, Sf[tid * 66 + j]);
        float s = 0.0f;
        for (int j = 0; j < 64; ++j) {
            const float e = __expf(Sf[tid * 66 + j] - m);
            Sf[tid * 66 + j] = e;
            s += e;
        }
        const float inv = 1.0f / s;
        for (int j = 0; j < 64; ++j) Ps[tid * 72 + j] = (bf16)(Sf[tid * 66 + j] * inv);
    }
    __syncthreads();

    // ---- O = P.V ----
    const v16bf p0 = load_a_frag<72>(Ps,      wv * 16, lane);
    const v16bf p1 = load_a_frag<72>(Ps + 32, wv * 16, lane);
    v8f oacc[4];
    const v8f vzero = {};
#pragma unroll
    for (int t = 0; t < 4; ++t) {
        oacc[t] = vzero;
        const v16bf bv0 = load_b_frag<72>(Vs,           t * 16, lane);
        const v16bf bv1 = load_b_frag<72>(Vs + 32 * 72, t * 16, lane);
        oacc[t] = WMMA_BF16(p0, bv0, oacc[t]);
        oacc[t] = WMMA_BF16(p1, bv1, oacc[t]);
    }
    __syncthreads();                      // everyone done reading Ps

    // stage O (bf16) in LDS, then coalesced uint4 store with reverse shift
#pragma unroll
    for (int t = 0; t < 4; ++t) {
#pragma unroll
        for (int j = 0; j < 8; ++j) {
            const int ig = wv * 16 + j + ((lane >> 4) << 3);
            const int dg = t * 16 + (lane & 15);
            Ps[ig * 72 + dg] = (bf16)oacc[t][j];
        }
    }
    __syncthreads();
    {
        const int i  = tid >> 1;
        const int d0 = (tid & 1) * 32;
        const int r = i >> 3, c = i & 7;
        const int y = (wy * 8 + r + shift) & 31;
        const int x = (wx * 8 + c + shift) & 31;
        uint4* dst = (uint4*)(o + ((size_t)b * 1024 + y * 32 + x) * 1024 + (size_t)head * 64 + d0);
        const uint4* src = (const uint4*)&Ps[i * 72 + d0];
#pragma unroll
        for (int u = 0; u < 4; ++u) dst[u] = src[u];
    }
}

// ---------------------------------------------------------------------------
extern "C" void kernel_launch(void* const* d_in, const int* in_sizes, int n_in,
                              void* d_out, int out_size, void* d_ws, size_t ws_size,
                              hipStream_t stream)
{
    const float* x          = (const float*)d_in[0];   // 64 x 1024 x 512
    const float* w_qkv      = (const float*)d_in[1];   // 512 x 3072
    const float* bias_table = (const float*)d_in[2];   // 225 x 16
    const float* w_out      = (const float*)d_in[3];   // 1024 x 512
    const float* b_out      = (const float*)d_in[4];   // 512
    float* out              = (float*)d_out;           // 64 x 1024 x 512

    bf16* qkv_ws  = (bf16*)d_ws;                       // 65536 x 3072 bf16
    bf16* attn_ws = qkv_ws + (size_t)65536 * 3072;     // 65536 x 1024 bf16

    const dim3 blk(128);

    // 1) qkv = x @ w_qkv  (fp32 in -> bf16 out), M=65536 N=3072 K=512
    gemm_wmma_kernel<false, false><<<dim3(3072 / 64, 65536 / 128), blk, 0, stream>>>(
        (const void*)x, w_qkv, nullptr, (void*)qkv_ws, 65536, 3072, 512);

    // 2) windowed attention: 1024 windows x 16 heads
    swin_attn_kernel<<<dim3(1024, 16), blk, 0, stream>>>(qkv_ws, bias_table, attn_ws);

    // 3) out = attn @ w_out + b_out  (bf16 in -> fp32 out), M=65536 N=512 K=1024
    gemm_wmma_kernel<true, true><<<dim3(512 / 64, 65536 / 128), blk, 0, stream>>>(
        (const void*)attn_ws, w_out, b_out, (void*)out, 65536, 512, 1024);
}
